// Block_730144441078
// MI455X (gfx1250) — matrix operand order repaired and checked by reading, not compiled, not run on hardware
//
#include <hip/hip_runtime.h>

// ---------------------------------------------------------------------------
// CDNA5 (gfx1250) fused transformer block, round 3.
// All GEMMs computed in TRANSPOSED form (out^T = B^T * A^T) so every WMMA
// D-tile store has M (VGPR axis) on the contiguous LDS/global axis ->
// single ds_store_b128 / global_store_b128 per tile. Weight matrices are
// pre-packed (f32 -> bf16) into WMMA *A-fragment* order in d_ws, so the
// A-side of every big GEMM streams from L2 with one 32B load per lane.
// Input x tile is DMA'd into LDS via the Tensor Data Mover.
// ---------------------------------------------------------------------------

typedef __attribute__((ext_vector_type(16))) __bf16 v16bf;
typedef __attribute__((ext_vector_type(8)))  __bf16 v8bf;
typedef __attribute__((ext_vector_type(8)))  float  v8f;
typedef __attribute__((ext_vector_type(4)))  float  v4f;
typedef __attribute__((ext_vector_type(4)))  unsigned int v4u;
typedef __attribute__((ext_vector_type(8)))  int    v8i;
typedef __attribute__((ext_vector_type(4)))  int    v4i;

#define WMMA_BF16(a, b, c) \
  __builtin_amdgcn_wmma_f32_16x16x32_bf16(false, (a), false, (b), (short)0, (c), false, false)

// TDM builtin arity differs between toolchains (probe-verified):
// ROCm 7.2 / clang-22 -> 5 args; amdgpu-toolchain / clang-23 -> 6 args.
#if defined(__clang_major__) && (__clang_major__ >= 23)
#define TDM_LOAD_TO_LDS(g0, g1, g2, g3) \
  __builtin_amdgcn_tensor_load_to_lds((g0), (g1), (g2), (g3), (v8i){0, 0, 0, 0, 0, 0, 0, 0}, 0)
#else
#define TDM_LOAD_TO_LDS(g0, g1, g2, g3) \
  __builtin_amdgcn_tensor_load_to_lds((g0), (g1), (g2), (g3), 0)
#endif

static __device__ inline v8f zero8() {
  v8f z = {0.f, 0.f, 0.f, 0.f, 0.f, 0.f, 0.f, 0.f};
  return z;
}

// A fragment (16x32 bf16) from row-major LDS: two ds_load_b128 per lane.
static __device__ inline v16bf load_A_lds(const __bf16* s, int ld, int lane) {
  const int m  = lane & 15;
  const int kb = (lane >> 4) << 3;  // 0 or 8
  const __bf16* p = s + m * ld + kb;
  const v8bf lo = *(const v8bf*)(p);
  const v8bf hi = *(const v8bf*)(p + 16);
  v16bf a;
#pragma unroll
  for (int j = 0; j < 8; ++j) { a[j] = lo[j]; a[8 + j] = hi[j]; }
  return a;
}

// B fragment (32x16) where logical B[k][n] = src[n*ld + k]: contiguous along
// k -> two ds_load_b128 per lane. All LDS-resident B operands use this.
static __device__ inline v16bf load_B_ldsT(const __bf16* s, int ld, int lane) {
  const int n  = lane & 15;
  const int k0 = (lane >> 4) << 3;  // 0 or 8
  const __bf16* p = s + n * ld + k0;
  const v8bf lo = *(const v8bf*)(p);
  const v8bf hi = *(const v8bf*)(p + 16);
  v16bf b;
#pragma unroll
  for (int j = 0; j < 8; ++j) { b[j] = lo[j]; b[8 + j] = hi[j]; }
  return b;
}

// Pre-packed fragment (A or B order): one 32B contiguous load per lane.
static __device__ inline v16bf load_frag_pk(const __bf16* base, int lane) {
  return *(const v16bf*)(base + (lane << 4));
}

// D tile store with M on the contiguous axis: dT[n*ldT + m], one
// ds_store_b128 per lane.
static __device__ inline void store_D_T(__bf16* dT, int ldT, int lane, v8f v) {
  const int n  = lane & 15;
  const int mh = (lane >> 4) << 3;
  v8bf t;
#pragma unroll
  for (int r = 0; r < 8; ++r) t[r] = (__bf16)v[r];
  *(v8bf*)(dT + n * ldT + mh) = t;
}

// ---------------------------------------------------------------------------
// Weight pre-pack kernels: f32 row-major KxN (row stride ld) -> bf16 in WMMA
// fragment order. One thread = one lane of one tile (writes 16 contig bf16).
// ---------------------------------------------------------------------------
__global__ void pack_B_kernel(const float* __restrict__ src, __bf16* __restrict__ dst,
                              int K, int N, int ld, int hs, int hd, int total) {
  const int t = blockIdx.x * blockDim.x + threadIdx.x;
  if (t >= total) return;
  const int lane   = t & 31;
  const int tile   = t >> 5;
  const int ntiles = N >> 4;
  const int ktiles = K >> 5;
  const int nt = tile % ntiles;
  const int kt = (tile / ntiles) % ktiles;
  const int h  = tile / (ntiles * ktiles);
  const float* s = src + (size_t)h * hs + (size_t)(kt * 32) * ld + nt * 16;
  __bf16* d = dst + (size_t)h * hd + ((size_t)(kt * ntiles + nt) * 32 + lane) * 16;
  const int n  = lane & 15;
  const int k0 = (lane >> 4) << 3;
  v16bf v;
#pragma unroll
  for (int j = 0; j < 8; ++j) v[j]     = (__bf16)s[(k0 + j) * ld + n];
#pragma unroll
  for (int j = 0; j < 8; ++j) v[8 + j] = (__bf16)s[(16 + k0 + j) * ld + n];
  *(v16bf*)d = v;
}

// A-fragment pack of W^T: A'[m][k] = W[k][m].  Tile index = mt*ktiles + kt.
__global__ void pack_AT_kernel(const float* __restrict__ src, __bf16* __restrict__ dst,
                               int K, int N, int ld, int hs, int hd, int total) {
  const int t = blockIdx.x * blockDim.x + threadIdx.x;
  if (t >= total) return;
  const int lane   = t & 31;
  const int tile   = t >> 5;
  const int ktiles = K >> 5;
  const int mtiles = N >> 4;
  const int kt = tile % ktiles;
  const int mt = (tile / ktiles) % mtiles;
  const int h  = tile / (ktiles * mtiles);
  const float* s = src + (size_t)h * hs + (size_t)(kt * 32) * ld + mt * 16;
  __bf16* d = dst + (size_t)h * hd + ((size_t)(mt * ktiles + kt) * 32 + lane) * 16;
  const int m  = lane & 15;
  const int kb = (lane >> 4) << 3;
  v16bf v;
#pragma unroll
  for (int j = 0; j < 8; ++j) v[j]     = (__bf16)s[(kb + j) * ld + m];
#pragma unroll
  for (int j = 0; j < 8; ++j) v[8 + j] = (__bf16)s[(16 + kb + j) * ld + m];
  *(v16bf*)d = v;
}

// ---------------------------------------------------------------------------
// Main fused transformer block kernel.
// ---------------------------------------------------------------------------
__global__ __launch_bounds__(256) void txblock_kernel(
    const float* __restrict__ X,
    const float* __restrict__ ln1g, const float* __restrict__ ln1bv,
    const __bf16* __restrict__ pQ, const __bf16* __restrict__ pK,
    const __bf16* __restrict__ pV, const __bf16* __restrict__ pP,
    const float* __restrict__ bp,
    const float* __restrict__ ln2g, const float* __restrict__ ln2bv,
    const __bf16* __restrict__ p1, const float* __restrict__ b1v,
    const __bf16* __restrict__ p2, const float* __restrict__ b2v,
    float* __restrict__ Out) {
  constexpr int T = 64, C = 256, D = 64, H = 4;

  __shared__ __align__(32) float  xs[T * C];   // 64 KB : x, later x + sa
  __shared__ __align__(32) __bf16 hb[T * C];   // 32 KB : LN1 out / LN2 out
  __shared__ __align__(32) __bf16 ab[T * C];   // 32 KB : att concat / ffn gate
  __shared__ __align__(32) __bf16 qh[T * D];   //  8 KB : Q (token-major)
  __shared__ __align__(32) __bf16 kh[T * D];   //  8 KB : K (token-major)
  __shared__ __align__(32) __bf16 vt[D * T];   //  8 KB : V transposed (d-major)
  __shared__ __align__(32) __bf16 pbuf[T * T]; //  8 KB : softmax probs P
  __shared__ float red2[T][2];
  __shared__ float rmax[T];
  __shared__ float rsum[T];

  const int b    = blockIdx.x;
  const int tid  = threadIdx.x;
  const int lane = tid & 31;
  const int w    = tid >> 5;
  const int mt   = w & 3;   // 4-way tile split
  const int nh   = w >> 2;  // 2-way half split

  const float* xg = X + (size_t)b * T * C;
  float*       og = Out + (size_t)b * T * C;

  if (tid == 0) {  // global_prefetch_b8: seed L2 with packed weights
    __builtin_prefetch(pQ, 0, 1); __builtin_prefetch(pK, 0, 1);
    __builtin_prefetch(pV, 0, 1); __builtin_prefetch(pP, 0, 1);
    __builtin_prefetch(p1, 0, 1); __builtin_prefetch(p2, 0, 1);
  }

  // ---- Tensor Data Mover: DMA the 64x256 f32 x-tile (contiguous) into LDS --
  if (w == 0) {
    const unsigned long long ga = (unsigned long long)(uintptr_t)xg;
    const unsigned int lds = (unsigned int)(uintptr_t)(&xs[0]);
    v4u g0;
    g0[0] = 1u;                                       // count=1 (valid user D#)
    g0[1] = lds;                                      // lds_addr
    g0[2] = (unsigned int)(ga & 0xffffffffu);         // global_addr[31:0]
    g0[3] = (unsigned int)((ga >> 32) & 0x01ffffffu)  // global_addr[56:32]
            | 0x80000000u;                            // type=2 ("image")
    v8i g1;
    g1[0] = 0x00020000;  // data_size=2 -> 4-byte elements
    g1[1] = 0x40000000;  // tensor_dim0[15:0] = 16384
    g1[2] = 0x00010000;  // tensor_dim1 = 1
    g1[3] = 0x40000000;  // tile_dim0 = 16384
    g1[4] = 1;           // tile_dim1 = 1
    g1[5] = 16384;       // tensor_dim0_stride
    g1[6] = 0;
    g1[7] = 0;
    const v4i gz = {0, 0, 0, 0};
    TDM_LOAD_TO_LDS(g0, g1, gz, gz);
    __builtin_amdgcn_s_wait_tensorcnt(0);
  }
  __syncthreads();

  // ------------------------- LayerNorm 1 (4 lanes / row) -------------------
  {
    const int row = tid >> 2, seg = (tid & 3) * 64;
    const float* xr = xs + row * C + seg;
    float s = 0.f, s2 = 0.f;
#pragma unroll
    for (int j4 = 0; j4 < 16; ++j4) {
      const v4f v = *(const v4f*)(xr + j4 * 4);
#pragma unroll
      for (int e = 0; e < 4; ++e) { s += v[e]; s2 += v[e] * v[e]; }
    }
    s  += __shfl_xor(s, 1, 32);  s  += __shfl_xor(s, 2, 32);
    s2 += __shfl_xor(s2, 1, 32); s2 += __shfl_xor(s2, 2, 32);
    const float mu  = s * (1.f / C);
    const float var = s2 * (1.f / C) - mu * mu;
    const float rs  = rsqrtf(var + 1e-6f);
#pragma unroll
    for (int j8 = 0; j8 < 8; ++j8) {
      v8bf o;
#pragma unroll
      for (int r = 0; r < 8; ++r) {
        const int c = seg + j8 * 8 + r;
        o[r] = (__bf16)((xs[row * C + c] - mu) * rs * ln1g[c] + ln1bv[c]);
      }
      *(v8bf*)(hb + row * C + seg + j8 * 8) = o;
    }
  }
  __syncthreads();

  // ------------------------- attention, one head at a time -----------------
  for (int h = 0; h < H; ++h) {
    // q^T = Wq^T h^T , k^T = Wk^T h^T  (A = packed W^T, B = h^T from hb)
    // v   = h Wv                        (A = hb, B = packed Wv)
    {
      v8f aq[2], ak[2], av[2];
#pragma unroll
      for (int j = 0; j < 2; ++j) { aq[j] = zero8(); ak[j] = zero8(); av[j] = zero8(); }
      const __bf16* aTq = pQ + h * 16384;   // A': 4 mtiles(d) * 8 ktiles(c) * 512
      const __bf16* aTk = pK + h * 16384;
      const __bf16* bVh = pV + h * 16384;   // B : 8 ktiles(c) * 4 ntiles(d) * 512
#pragma unroll
      for (int kk = 0; kk < C / 32; ++kk) {
        const v16bf aQ = load_frag_pk(aTq + (mt * 8 + kk) * 512, lane);
        const v16bf aK = load_frag_pk(aTk + (mt * 8 + kk) * 512, lane);
        const v16bf aV = load_A_lds(hb + mt * 16 * C + kk * 32, C, lane);
#pragma unroll
        for (int j = 0; j < 2; ++j) {
          const int nt = nh * 2 + j;
          const v16bf bT = load_B_ldsT(hb + nt * 16 * C + kk * 32, C, lane); // h^T
          aq[j] = WMMA_BF16(aQ, bT, aq[j]);
          ak[j] = WMMA_BF16(aK, bT, ak[j]);
          av[j] = WMMA_BF16(aV, load_frag_pk(bVh + (kk * 4 + nt) * 512, lane), av[j]);
        }
      }
#pragma unroll
      for (int j = 0; j < 2; ++j) {
        const int t0 = (nh * 2 + j) * 16;
        store_D_T(qh + t0 * D + mt * 16, D, lane, aq[j]);  // qh[t*64 + d]
        store_D_T(kh + t0 * D + mt * 16, D, lane, ak[j]);  // kh[t*64 + d]
        store_D_T(vt + t0 * T + mt * 16, T, lane, av[j]);  // vt[d*64 + s]
      }
    }
    __syncthreads();

    // S^T = K Q^T : lane -> t (all of a row in one lane), VGPR -> s.
    // wave w: t-tile mt, s-tiles (nh*2 + {0,1}).
    float sv[2][8];
    {
      v8f acc[2] = {zero8(), zero8()};
#pragma unroll
      for (int kk = 0; kk < 2; ++kk) {
        const v16bf bQ = load_B_ldsT(qh + mt * 16 * D + kk * 32, D, lane); // B[k=d][n=t]
#pragma unroll
        for (int j = 0; j < 2; ++j) {
          const v16bf aK = load_A_lds(kh + (nh * 2 + j) * 16 * D + kk * 32, D, lane);
          acc[j] = WMMA_BF16(aK, bQ, acc[j]);
        }
      }
      const int tcol = mt * 16 + (lane & 15);
      const int sb0  = (lane >> 4) << 3;
#pragma unroll
      for (int j = 0; j < 2; ++j) {
        const int s0 = (nh * 2 + j) * 16 + sb0;
#pragma unroll
        for (int r = 0; r < 8; ++r)
          sv[j][r] = (s0 + r <= tcol) ? acc[j][r] * 0.125f : -3.0e38f;
      }
    }

    // per-lane max over 16 s-values, pair lanes via xor(16), then LDS combine
    {
      float pm = sv[0][0];
#pragma unroll
      for (int r = 1; r < 8; ++r) pm = fmaxf(pm, sv[0][r]);
#pragma unroll
      for (int r = 0; r < 8; ++r) pm = fmaxf(pm, sv[1][r]);
      pm = fmaxf(pm, __shfl_xor(pm, 16, 32));
      if (lane < 16) red2[mt * 16 + lane][nh] = pm;
    }
    __syncthreads();
    if (tid < T) rmax[tid] = fmaxf(red2[tid][0], red2[tid][1]);
    __syncthreads();

    float ev[2][8];
    {
      const float m = rmax[mt * 16 + (lane & 15)];
      float ps = 0.f;
#pragma unroll
      for (int j = 0; j < 2; ++j)
#pragma unroll
        for (int r = 0; r < 8; ++r) { ev[j][r] = __expf(sv[j][r] - m); ps += ev[j][r]; }
      ps += __shfl_xor(ps, 16, 32);
      if (lane < 16) red2[mt * 16 + lane][nh] = ps;
    }
    __syncthreads();
    if (tid < T) rsum[tid] = red2[tid][0] + red2[tid][1];
    __syncthreads();

    // P store: pbuf[t*64 + s] with s contiguous -> vector stores
    {
      const float inv = 1.f / rsum[mt * 16 + (lane & 15)];
#pragma unroll
      for (int j = 0; j < 2; ++j) {
        v8f pv;
#pragma unroll
        for (int r = 0; r < 8; ++r) pv[r] = ev[j][r] * inv;
        store_D_T(pbuf + mt * 16 * T + (nh * 2 + j) * 16, T, lane, pv);
      }
    }
    __syncthreads();

    // att^T = V^T P^T : A = vt (d-major), B[k=s][n=t] = pbuf[t*64+s].
    // wave w: d-tile mt, t-tiles (nh*2 + {0,1}).
#pragma unroll
    for (int j = 0; j < 2; ++j) {
      v8f acc = zero8();
#pragma unroll
      for (int kk = 0; kk < 2; ++kk) {
        const v16bf a  = load_A_lds(vt + mt * 16 * T + kk * 32, T, lane);
        const v16bf bb = load_B_ldsT(pbuf + (nh * 2 + j) * 16 * T + kk * 32, T, lane);
        acc = WMMA_BF16(a, bb, acc);
      }
      // ab[t*C + h*64 + d], d contiguous
      store_D_T(ab + (nh * 2 + j) * 16 * C + h * D + mt * 16, C, lane, acc);
    }
    __syncthreads();
  }

  // -------- sa^T = Wp^T att^T ; x += sa + bproj (vector f32 RMW) -----------
  {
    v8f acc[2][4];
#pragma unroll
    for (int i = 0; i < 2; ++i)
#pragma unroll
      for (int tj = 0; tj < 4; ++tj) acc[i][tj] = zero8();
#pragma unroll
    for (int kk = 0; kk < 8; ++kk) {
      v16bf aA[2];
#pragma unroll
      for (int i = 0; i < 2; ++i)
        aA[i] = load_frag_pk(pP + ((w * 2 + i) * 8 + kk) * 512, lane);
#pragma unroll
      for (int tj = 0; tj < 4; ++tj) {
        const v16bf bb = load_B_ldsT(ab + tj * 16 * C + kk * 32, C, lane);
#pragma unroll
        for (int i = 0; i < 2; ++i) acc[i][tj] = WMMA_BF16(aA[i], bb, acc[i][tj]);
      }
    }
    const int mhx = (lane >> 4) << 3;
#pragma unroll
    for (int i = 0; i < 2; ++i) {
      const int c0 = (w * 2 + i) * 16 + mhx;
      const v4f bp0 = *(const v4f*)(bp + c0);
      const v4f bp1 = *(const v4f*)(bp + c0 + 4);
#pragma unroll
      for (int tj = 0; tj < 4; ++tj) {
        const int t = tj * 16 + (lane & 15);
        float* xr = xs + t * C + c0;
        v4f x0 = *(v4f*)(xr);
        v4f x1 = *(v4f*)(xr + 4);
#pragma unroll
        for (int e = 0; e < 4; ++e) {
          x0[e] += acc[i][tj][e] + bp0[e];
          x1[e] += acc[i][tj][4 + e] + bp1[e];
        }
        *(v4f*)(xr) = x0;
        *(v4f*)(xr + 4) = x1;
      }
    }
  }
  __syncthreads();

  // ------------------------- LayerNorm 2 -----------------------------------
  {
    const int row = tid >> 2, seg = (tid & 3) * 64;
    const float* xr = xs + row * C + seg;
    float s = 0.f, s2 = 0.f;
#pragma unroll
    for (int j4 = 0; j4 < 16; ++j4) {
      const v4f v = *(const v4f*)(xr + j4 * 4);
#pragma unroll
      for (int e = 0; e < 4; ++e) { s += v[e]; s2 += v[e] * v[e]; }
    }
    s  += __shfl_xor(s, 1, 32);  s  += __shfl_xor(s, 2, 32);
    s2 += __shfl_xor(s2, 1, 32); s2 += __shfl_xor(s2, 2, 32);
    const float mu  = s * (1.f / C);
    const float var = s2 * (1.f / C) - mu * mu;
    const float rs  = rsqrtf(var + 1e-6f);
#pragma unroll
    for (int j8 = 0; j8 < 8; ++j8) {
      v8bf o;
#pragma unroll
      for (int r = 0; r < 8; ++r) {
        const int c = seg + j8 * 8 + r;
        o[r] = (__bf16)((xs[row * C + c] - mu) * rs * ln2g[c] + ln2bv[c]);
      }
      *(v8bf*)(hb + row * C + seg + j8 * 8) = o;
    }
  }
  __syncthreads();

  // ------------------------- FFN (transposed, 4 K-chunks) ------------------
  {
    v8f facc[2][4];
#pragma unroll
    for (int i = 0; i < 2; ++i)
#pragma unroll
      for (int tj = 0; tj < 4; ++tj) facc[i][tj] = zero8();

    for (int ch = 0; ch < 4; ++ch) {
      // gate^T = W1^T h2^T ; relu(+b1); store token-major into ab
      v8f g[2][4];
#pragma unroll
      for (int i = 0; i < 2; ++i)
#pragma unroll
        for (int tj = 0; tj < 4; ++tj) g[i][tj] = zero8();
#pragma unroll
      for (int kk = 0; kk < 8; ++kk) {
        v16bf aA[2];
#pragma unroll
        for (int i = 0; i < 2; ++i)
          aA[i] = load_frag_pk(p1 + ((size_t)((ch * 16 + w * 2 + i) * 8 + kk)) * 512, lane);
#pragma unroll
        for (int tj = 0; tj < 4; ++tj) {
          const v16bf bb = load_B_ldsT(hb + tj * 16 * C + kk * 32, C, lane);
#pragma unroll
          for (int i = 0; i < 2; ++i) g[i][tj] = WMMA_BF16(aA[i], bb, g[i][tj]);
        }
      }
      __syncthreads();  // prior chunk's readers of ab are done
      const int mhx = (lane >> 4) << 3;
#pragma unroll
      for (int i = 0; i < 2; ++i) {
        const int m0 = (w * 2 + i) * 16 + mhx;           // hidden-local
        const v4f bg0 = *(const v4f*)(b1v + ch * 256 + m0);
        const v4f bg1 = *(const v4f*)(b1v + ch * 256 + m0 + 4);
#pragma unroll
        for (int tj = 0; tj < 4; ++tj) {
          v8f t;
#pragma unroll
          for (int e = 0; e < 4; ++e) {
            t[e]     = fmaxf(g[i][tj][e] + bg0[e], 0.f);
            t[4 + e] = fmaxf(g[i][tj][4 + e] + bg1[e], 0.f);
          }
          store_D_T(ab + tj * 16 * C + (w * 2 + i) * 16, C, lane, t);
        }
      }
      __syncthreads();
      // ff^T += W2^T gate^T
#pragma unroll
      for (int kk = 0; kk < 8; ++kk) {
        v16bf aA[2];
#pragma unroll
        for (int i = 0; i < 2; ++i)
          aA[i] = load_frag_pk(p2 + ((size_t)((w * 2 + i) * 32 + ch * 8 + kk)) * 512, lane);
#pragma unroll
        for (int tj = 0; tj < 4; ++tj) {
          const v16bf bb = load_B_ldsT(ab + tj * 16 * C + kk * 32, C, lane);
#pragma unroll
          for (int i = 0; i < 2; ++i) facc[i][tj] = WMMA_BF16(aA[i], bb, facc[i][tj]);
        }
      }
    }

    // out = (x + sa) + ff + b2 : vector global stores (8 contig channels/lane)
    const int mhx = (lane >> 4) << 3;
#pragma unroll
    for (int i = 0; i < 2; ++i) {
      const int c0 = (w * 2 + i) * 16 + mhx;
      const v4f b20 = *(const v4f*)(b2v + c0);
      const v4f b21 = *(const v4f*)(b2v + c0 + 4);
#pragma unroll
      for (int tj = 0; tj < 4; ++tj) {
        const int t = tj * 16 + (lane & 15);
        const float* xr = xs + t * C + c0;
        v4f o0 = *(const v4f*)(xr);
        v4f o1 = *(const v4f*)(xr + 4);
#pragma unroll
        for (int e = 0; e < 4; ++e) {
          o0[e] += facc[i][tj][e] + b20[e];
          o1[e] += facc[i][tj][4 + e] + b21[e];
        }
        *(v4f*)(og + t * C + c0)     = o0;
        *(v4f*)(og + t * C + c0 + 4) = o1;
      }
    }
  }
}

extern "C" void kernel_launch(void* const* d_in, const int* in_sizes, int n_in,
                              void* d_out, int out_size, void* d_ws, size_t ws_size,
                              hipStream_t stream) {
  const float* X    = (const float*)d_in[0];
  const float* g1   = (const float*)d_in[1];
  const float* bln1 = (const float*)d_in[2];
  const float* Wq   = (const float*)d_in[3];
  const float* Wk   = (const float*)d_in[4];
  const float* Wv   = (const float*)d_in[5];
  const float* Wp   = (const float*)d_in[6];
  const float* bpj  = (const float*)d_in[7];
  const float* g2   = (const float*)d_in[8];
  const float* bln2 = (const float*)d_in[9];
  const float* W1   = (const float*)d_in[10];
  const float* b1   = (const float*)d_in[11];
  const float* W2   = (const float*)d_in[12];
  const float* b2   = (const float*)d_in[13];
  float* out = (float*)d_out;

  // Packed-weight workspace layout (bf16 elements, 1.5 MB total):
  __bf16* ws = (__bf16*)d_ws;
  __bf16* pQ  = ws;             // A-pack Wq^T : 4 heads * 4mt*8kt*512
  __bf16* pK  = ws + 65536;     // A-pack Wk^T
  __bf16* pV  = ws + 131072;    // B-pack Wv
  __bf16* pP  = ws + 196608;    // A-pack Wp^T : 16mt*8kt*512
  __bf16* pW1 = ws + 262144;    // A-pack W1^T : 64mt*8kt*512
  __bf16* pW2 = ws + 524288;    // A-pack W2^T : 16mt*32kt*512

  auto packA = [&](const float* src, __bf16* dst, int K, int N, int ld,
                   int hs, int hd, int nH) {
    const int total = nH * (K / 32) * (N / 16) * 32;
    hipLaunchKernelGGL(pack_AT_kernel, dim3((total + 255) / 256), dim3(256), 0, stream,
                       src, dst, K, N, ld, hs, hd, total);
  };
  auto packB = [&](const float* src, __bf16* dst, int K, int N, int ld,
                   int hs, int hd, int nH) {
    const int total = nH * (K / 32) * (N / 16) * 32;
    hipLaunchKernelGGL(pack_B_kernel, dim3((total + 255) / 256), dim3(256), 0, stream,
                       src, dst, K, N, ld, hs, hd, total);
  };
  packA(Wq, pQ, 256, 64, 64, 16384, 16384, 4);
  packA(Wk, pK, 256, 64, 64, 16384, 16384, 4);
  packB(Wv, pV, 256, 64, 64, 16384, 16384, 4);
  packA(Wp, pP, 256, 256, 256, 0, 0, 1);
  packA(W1, pW1, 256, 1024, 1024, 0, 0, 1);
  packA(W2, pW2, 1024, 256, 256, 0, 0, 1);

  const int B = in_sizes[0] / (64 * 256);  // 4096 for the reference shapes
  hipLaunchKernelGGL(txblock_kernel, dim3(B), dim3(256), 0, stream,
                     X, g1, bln1, pQ, pK, pV, pP, bpj, g2, bln2,
                     pW1, b1, pW2, b2, out);
}
